// CrossAttention_80444737454805
// MI455X (gfx1250) — matrix operand order repaired and checked by reading, not compiled, not run on hardware
//
#include <hip/hip_runtime.h>
#include <hip/hip_bf16.h>

// ---------- types ----------
typedef __attribute__((ext_vector_type(16))) __bf16 v16bf;
typedef __attribute__((ext_vector_type(8)))  float  v8f;
typedef __attribute__((ext_vector_type(4)))  unsigned int u32x4;

union Frag16 { v16bf v; u32x4 q[2]; __bf16 e[16]; };

__device__ __forceinline__ v16bf load_frag2(const __bf16* p0, const __bf16* p1) {
  Frag16 f;
  f.q[0] = *reinterpret_cast<const u32x4*>(p0);
  f.q[1] = *reinterpret_cast<const u32x4*>(p1);
  return f.v;
}

__device__ __forceinline__ v8f wmma_bf16(v16bf a, v16bf b, v8f c) {
  // D = A(16x32 bf16) * B(32x16 bf16) + C(16x16 f32)
  return __builtin_amdgcn_wmma_f32_16x16x32_bf16(
      /*neg_a=*/false, a, /*neg_b=*/false, b,
      /*c_mod=*/(short)0, c, /*reuse_a=*/false, /*reuse_b=*/false);
}

__device__ __forceinline__ v8f zero_v8f() {
  v8f z = {0.f, 0.f, 0.f, 0.f, 0.f, 0.f, 0.f, 0.f};
  return z;
}

// CDNA5 async copy: global -> LDS, 16 bytes per lane, tracked by ASYNCcnt.
// Generic LDS pointer's low 32 bits are the LDS byte address (aperture rule).
__device__ __forceinline__ void async_load_b128(const void* gsrc, void* lds_dst) {
  unsigned lds = (unsigned)(unsigned long long)lds_dst;
  unsigned long long g = (unsigned long long)gsrc;
  asm volatile("global_load_async_to_lds_b128 %0, %1, off"
               :: "v"(lds), "v"(g) : "memory");
}
__device__ __forceinline__ void wait_async0() {
  asm volatile("s_wait_asynccnt 0" ::: "memory");
}

// ---------- fp32 -> bf16 conversion ----------
__global__ void cvt_f32_bf16(const float* __restrict__ in, __bf16* __restrict__ out,
                             long long n) {
  long long i = (long long)blockIdx.x * blockDim.x + threadIdx.x;
  long long st = (long long)gridDim.x * blockDim.x;
  for (; i < n; i += st) out[i] = (__bf16)in[i];
}

// ---------- generic bf16 GEMM: C[MxN] = A[MxK] @ B[KxN] (+bias) ----------
// Block tile 128x128, 256 threads = 8 waves (2 in M x 4 in N),
// each wave: 64x32 via 4x2 grid of v_wmma_f32_16x16x32_bf16, K-step 32.
// A tile staged via async global->LDS; B tile transposed through VGPRs.
template <bool OUT_BF16, bool HAS_BIAS>
__global__ __launch_bounds__(256)
void gemm_bf16_k(const __bf16* __restrict__ A, const __bf16* __restrict__ B,
                 void* __restrict__ Cout, const float* __restrict__ bias,
                 int M, int N, int K) {
  constexpr int LDSLD = 40;  // bf16 elems per LDS row (80B, 16B aligned, padded)
  __shared__ __bf16 As[128 * LDSLD];  // A tile, row-major [row][k]
  __shared__ __bf16 Bt[128 * LDSLD];  // B tile transposed [n][k]

  const int tid  = threadIdx.x;
  const int lane = tid & 31;
  const int wv   = tid >> 5;
  const int wm   = wv & 1;   // wave pos in M (2)
  const int wn   = wv >> 1;  // wave pos in N (4)
  const int lr   = lane & 15;
  const int half = lane >> 4;

  const int rowBase = blockIdx.y * 128;
  const int colBase = blockIdx.x * 128;

  const int ar   = tid >> 1;          // A stage: row
  const int aseg = (tid & 1) << 4;    // A stage: 16-elem segment
  const int bk   = tid >> 3;          // B stage: k row
  const int bseg = (tid & 7) << 4;    // B stage: 16-col segment

  v8f c[4][2];
#pragma unroll
  for (int mi = 0; mi < 4; ++mi)
#pragma unroll
    for (int ni = 0; ni < 2; ++ni) c[mi][ni] = zero_v8f();

  for (int k0 = 0; k0 < K; k0 += 32) {
    // stage A tile asynchronously: 128 rows x 32 k (2x b128 per thread)
    {
      const __bf16* src = A + (size_t)(rowBase + ar) * K + k0 + aseg;
      async_load_b128(src,     &As[ar * LDSLD + aseg]);
      async_load_b128(src + 8, &As[ar * LDSLD + aseg + 8]);
    }
    // warm L2 for next K-step while we transpose-stage B
    if (k0 + 32 < K) {
      __builtin_prefetch(A + (size_t)(rowBase + ar) * K + k0 + 32 + aseg, 0, 0);
      __builtin_prefetch(B + (size_t)(k0 + 32 + bk) * N + colBase + bseg, 0, 0);
    }
    // stage B tile transposed: coalesced global read, strided LDS write
    {
      const __bf16* src = B + (size_t)(k0 + bk) * N + colBase + bseg;
      u32x4 b0 = *reinterpret_cast<const u32x4*>(src);
      u32x4 b1 = *reinterpret_cast<const u32x4*>(src + 8);
      const __bf16* e0 = reinterpret_cast<const __bf16*>(&b0);
      const __bf16* e1 = reinterpret_cast<const __bf16*>(&b1);
#pragma unroll
      for (int i = 0; i < 8; ++i) Bt[(bseg + i) * LDSLD + bk] = e0[i];
#pragma unroll
      for (int i = 0; i < 8; ++i) Bt[(bseg + 8 + i) * LDSLD + bk] = e1[i];
    }
    wait_async0();
    __syncthreads();

    // A fragments: row = base + lane%16; elems 0-7 @ K=half*8, 8-15 @ K=16+half*8
    v16bf af[4];
#pragma unroll
    for (int mi = 0; mi < 4; ++mi) {
      const __bf16* p = &As[(wm * 64 + mi * 16 + lr) * LDSLD + half * 8];
      af[mi] = load_frag2(p, p + 16);
    }
    // B fragments: col = base + lane%16; elems 0-7 @ K=half*8, 8-15 @ K=16+half*8
    v16bf bfv[2];
#pragma unroll
    for (int ni = 0; ni < 2; ++ni) {
      const __bf16* p = &Bt[(wn * 32 + ni * 16 + lr) * LDSLD + half * 8];
      bfv[ni] = load_frag2(p, p + 16);
    }
#pragma unroll
    for (int mi = 0; mi < 4; ++mi)
#pragma unroll
      for (int ni = 0; ni < 2; ++ni)
        c[mi][ni] = wmma_bf16(af[mi], bfv[ni], c[mi][ni]);
    __syncthreads();
  }

  // C layout: vgpr i -> row = half*8 + i, col = lane%16
#pragma unroll
  for (int mi = 0; mi < 4; ++mi)
#pragma unroll
    for (int ni = 0; ni < 2; ++ni)
#pragma unroll
      for (int i = 0; i < 8; ++i) {
        const int row = rowBase + wm * 64 + mi * 16 + half * 8 + i;
        const int col = colBase + wn * 32 + ni * 16 + lr;
        float v = c[mi][ni][i];
        if constexpr (HAS_BIAS) v += bias[col];
        if constexpr (OUT_BF16)
          reinterpret_cast<__bf16*>(Cout)[(size_t)row * N + col] = (__bf16)v;
        else
          reinterpret_cast<float*>(Cout)[(size_t)row * N + col] = v;
      }
}

// ---------- flash attention (bf16 WMMA, f32 accum) ----------
// grid (n/64, h, b); block 128 = 4 waves; wave w owns q rows n0+16w..+15.
__global__ __launch_bounds__(128)
void attn_k(const __bf16* __restrict__ qb,   // [b*1024, 1024]
            const __bf16* __restrict__ kvb,  // [b*4096, 2048] (k | v)
            const float* __restrict__ docsim,  // [b, 4]
            const float* __restrict__ betap,
            __bf16* __restrict__ aob) {      // [b*1024, 1024]
  constexpr int D = 1024, HD = 128, L = 4096, KVLD = 2048, NTOK = 1024;
  constexpr int LDSLD = 40;
  __shared__ __bf16 Vt[HD * LDSLD];      // V chunk transposed: [hd][32]
  __shared__ __bf16 Ps[4][16 * LDSLD];   // per-wave P tile [16][32]

  const int b = blockIdx.z, h = blockIdx.y;
  const int n0 = blockIdx.x * 64;
  const int tid = threadIdx.x;
  const int w = tid >> 5, lane = tid & 31, lr = lane & 15, half = lane >> 4;

  const float scale = 0.03125f;  // 1024^-0.5
  const float beta = betap[0];

  // q A-fragments (hd=128 -> 4 K-chunks of 32)
  const __bf16* qrow = qb + (size_t)(b * NTOK + n0 + w * 16 + lr) * D + h * HD;
  v16bf qf[4];
#pragma unroll
  for (int kc = 0; kc < 4; ++kc) {
    const __bf16* p = qrow + kc * 32 + half * 8;
    qf[kc] = load_frag2(p, p + 16);
  }

  v8f o[8];
  float m_[8], l_[8];
#pragma unroll
  for (int a = 0; a < 8; ++a) o[a] = zero_v8f();
#pragma unroll
  for (int i = 0; i < 8; ++i) { m_[i] = -1e30f; l_[i] = 0.f; }

  for (int j0 = 0; j0 < L; j0 += 32) {
    // stage V^T chunk (shared by all 4 waves): coalesced read, transposed write
#pragma unroll
    for (int r0 = 0; r0 < 32; r0 += 8) {
      const int r = r0 + (tid >> 4);
      const int cseg = (tid & 15) * 8;
      const __bf16* src =
          kvb + (size_t)(b * L + j0 + r) * KVLD + D + h * HD + cseg;
      u32x4 t = *reinterpret_cast<const u32x4*>(src);
      const __bf16* e = reinterpret_cast<const __bf16*>(&t);
#pragma unroll
      for (int i = 0; i < 8; ++i) Vt[(cseg + i) * LDSLD + r] = e[i];
    }
    __syncthreads();

    // dots = q @ k^T : k rows are B-fragments straight from global
    v8f c0 = zero_v8f(), c1 = zero_v8f();
    {
      const __bf16* k0p = kvb + (size_t)(b * L + j0 + lr) * KVLD + h * HD;
      const __bf16* k1p = k0p + (size_t)16 * KVLD;
#pragma unroll
      for (int kc = 0; kc < 4; ++kc) {
        const __bf16* p0 = k0p + kc * 32 + half * 8;
        const __bf16* p1 = k1p + kc * 32 + half * 8;
        c0 = wmma_bf16(qf[kc], load_frag2(p0, p0 + 16), c0);
        c1 = wmma_bf16(qf[kc], load_frag2(p1, p1 + 16), c1);
      }
    }

    // scale + per-doc similarity bias (doc = column >> 10)
    const float sim0 = docsim[b * 4 + ((j0 + lr) >> 10)] * beta;
    const float sim1 = docsim[b * 4 + ((j0 + 16 + lr) >> 10)] * beta;
    float s0[8], s1[8];
#pragma unroll
    for (int i = 0; i < 8; ++i) {
      s0[i] = c0[i] * scale + sim0;
      s1[i] = c1[i] * scale + sim1;
    }

    // online softmax: row r's 16 cols live in one 16-lane half -> xor<=8 reduce
    float cm[8];
#pragma unroll
    for (int i = 0; i < 8; ++i) cm[i] = fmaxf(s0[i], s1[i]);
#pragma unroll
    for (int off = 1; off < 16; off <<= 1)
#pragma unroll
      for (int i = 0; i < 8; ++i) cm[i] = fmaxf(cm[i], __shfl_xor(cm[i], off, 32));

    float alpha[8], p0v[8], p1v[8], rs[8];
#pragma unroll
    for (int i = 0; i < 8; ++i) {
      const float nm = fmaxf(m_[i], cm[i]);
      alpha[i] = __expf(m_[i] - nm);
      p0v[i] = __expf(s0[i] - nm);
      p1v[i] = __expf(s1[i] - nm);
      rs[i] = p0v[i] + p1v[i];
      m_[i] = nm;
    }
#pragma unroll
    for (int off = 1; off < 16; off <<= 1)
#pragma unroll
      for (int i = 0; i < 8; ++i) rs[i] += __shfl_xor(rs[i], off, 32);
#pragma unroll
    for (int i = 0; i < 8; ++i) l_[i] = l_[i] * alpha[i] + rs[i];
#pragma unroll
    for (int a = 0; a < 8; ++a)
#pragma unroll
      for (int i = 0; i < 8; ++i) o[a][i] *= alpha[i];

    // C-layout -> A-layout via per-wave LDS round trip
    __bf16* pr = &Ps[w][0];
#pragma unroll
    for (int i = 0; i < 8; ++i) {
      pr[(half * 8 + i) * LDSLD + lr] = (__bf16)p0v[i];
      pr[(half * 8 + i) * LDSLD + 16 + lr] = (__bf16)p1v[i];
    }
    asm volatile("s_wait_dscnt 0" ::: "memory");
    const __bf16* pp = &pr[lr * LDSLD + half * 8];
    v16bf pf = load_frag2(pp, pp + 16);

    // o += P(16x32) @ V(32x128): V^T rows are B-fragments
#pragma unroll
    for (int a = 0; a < 8; ++a) {
      const __bf16* vp = &Vt[(a * 16 + lr) * LDSLD + half * 8];
      o[a] = wmma_bf16(pf, load_frag2(vp, vp + 16), o[a]);
    }
    __syncthreads();
  }

  // normalize and store bf16 for the output projection GEMM
#pragma unroll
  for (int i = 0; i < 8; ++i) {
    const float inv = 1.0f / l_[i];
    const int row = n0 + w * 16 + half * 8 + i;
#pragma unroll
    for (int a = 0; a < 8; ++a)
      aob[(size_t)(b * NTOK + row) * D + h * HD + a * 16 + lr] =
          (__bf16)(o[a][i] * inv);
  }
}

// ---------- launcher ----------
extern "C" void kernel_launch(void* const* d_in, const int* in_sizes, int n_in,
                              void* d_out, int out_size, void* d_ws, size_t ws_size,
                              hipStream_t stream) {
  const float* x    = (const float*)d_in[0];  // [2,1024,1024]
  const float* ctx  = (const float*)d_in[1];  // [2,4,1024,1024]
  const float* dsim = (const float*)d_in[2];  // [2,4]
  const float* Wq   = (const float*)d_in[3];  // [1024,1024]
  const float* Wkv  = (const float*)d_in[4];  // [1024,2048]
  const float* beta = (const float*)d_in[5];  // scalar
  const float* Wout = (const float*)d_in[6];  // [1024,1024]
  const float* bout = (const float*)d_in[7];  // [1024]
  float* out = (float*)d_out;                 // [2,1024,1024] fp32

  char* wp = (char*)d_ws;
  auto take = [&](size_t elems) {
    __bf16* p = (__bf16*)wp;
    wp += elems * sizeof(__bf16);
    return p;
  };
  __bf16* xb   = take(2048ull * 1024);  // x bf16
  __bf16* cb   = take(8192ull * 1024);  // ctx bf16
  __bf16* wqb  = take(1024ull * 1024);
  __bf16* wkvb = take(1024ull * 2048);
  __bf16* wob  = take(1024ull * 1024);
  __bf16* qbuf = take(2048ull * 1024);  // q projection
  __bf16* kvb  = take(8192ull * 2048);  // kv projection
  __bf16* aob  = take(2048ull * 1024);  // attention output

  auto cvt = [&](const float* src, __bf16* dst, size_t n) {
    int blocks = (int)((n + 256ull * 8 - 1) / (256ull * 8));
    cvt_f32_bf16<<<blocks, 256, 0, stream>>>(src, dst, (long long)n);
  };
  cvt(x,    xb,   2048ull * 1024);
  cvt(ctx,  cb,   8192ull * 1024);
  cvt(Wq,   wqb,  1024ull * 1024);
  cvt(Wkv,  wkvb, 1024ull * 2048);
  cvt(Wout, wob,  1024ull * 1024);

  // q = x @ Wq : [2048,1024] = [2048,1024] @ [1024,1024]
  gemm_bf16_k<true, false><<<dim3(1024 / 128, 2048 / 128), 256, 0, stream>>>(
      xb, wqb, qbuf, nullptr, 2048, 1024, 1024);
  // kv = ctx @ Wkv : [8192,2048] = [8192,1024] @ [1024,2048]
  gemm_bf16_k<true, false><<<dim3(2048 / 128, 8192 / 128), 256, 0, stream>>>(
      cb, wkvb, kvb, nullptr, 8192, 2048, 1024);
  // attention
  attn_k<<<dim3(16, 8, 2), 128, 0, stream>>>(qbuf, kvb, dsim, beta, aob);
  // out = attn @ Wout + bout (fp32 out)
  gemm_bf16_k<false, true><<<dim3(1024 / 128, 2048 / 128), 256, 0, stream>>>(
      aob, wob, out, bout, 2048, 1024, 1024);
}
